// GlobalPrototype_68994354643493
// MI455X (gfx1250) — matrix operand order repaired and checked by reading, not compiled, not run on hardware
//
#include <hip/hip_runtime.h>
#include <hip/hip_bf16.h>
#include <stdint.h>

// ---------------------------------------------------------------------------
// Problem constants (from the reference): features (8,512,128,128) fp32,
// prototypes (150,512) fp32, output (8,150,128,128) fp32 cosine similarity.
// ---------------------------------------------------------------------------
#define BATCH 8
#define CDIM  512
#define HWDIM 16384          // 128*128
#define KOUT  150
#define KPAD  160            // 10 WMMA n-tiles of 16
#define MBLK  128            // pixels per workgroup
#define CCHUNK 32            // K-dim per WMMA (bf16 16x16x32)
#define NCHUNKS (CDIM / CCHUNK) // 16
#define EPS 1e-8f

typedef __attribute__((ext_vector_type(4)))  unsigned int u32x4;
typedef __attribute__((ext_vector_type(4)))  float        f32x4;
typedef __attribute__((ext_vector_type(8)))  float        v8f;
typedef __attribute__((ext_vector_type(8)))  __bf16       v8bf;
typedef __attribute__((ext_vector_type(16))) __bf16       v16bf;

union BF16Frag { v8bf h[2]; v16bf full; };

// ---------------------------------------------------------------------------
// Kernel 1: prototypes fp32 (150,512) -> bf16 (160,512) in ws (zero padded),
// plus per-prototype L2 norm. Native bf16 convert (v_cvt_*bf16*).
// ---------------------------------------------------------------------------
__global__ __launch_bounds__(128)
void proto_prep(const float* __restrict__ proto,
                __bf16* __restrict__ wsP,
                float* __restrict__ wsN) {
    const int k = blockIdx.x;          // 0..159
    const int t = threadIdx.x;         // 0..127
    __shared__ float red[128];

    float sq = 0.f;
    #pragma unroll
    for (int i = 0; i < 4; ++i) {
        const int c = t + i * 128;
        float v = 0.f;
        if (k < KOUT) v = proto[k * CDIM + c];
        sq += v * v;
        wsP[k * CDIM + c] = (__bf16)v;
    }
    red[t] = sq;
    __syncthreads();
    #pragma unroll
    for (int s = 64; s > 0; s >>= 1) {
        if (t < s) red[t] += red[t + s];
        __syncthreads();
    }
    if (t == 0) wsN[k] = (k < KOUT) ? sqrtf(red[0]) : 1.0f;
}

// ---------------------------------------------------------------------------
// Kernel 2: per block: 128 pixels x 160 prototypes GEMM via bf16 WMMA with
// fused sum-of-squares (feature norm), double-buffered LDS pipeline
// (one barrier per K-chunk), then normalize + store.
// ---------------------------------------------------------------------------
__global__ __launch_bounds__(256)
void cos_sim_gemm(const float* __restrict__ feat,
                  const __bf16* __restrict__ wsP,
                  const float* __restrict__ wsN,
                  float* __restrict__ out) {
    __shared__ alignas(16) __bf16 ldsA[2][MBLK][CCHUNK];   // 2 x 8 KB  ([m][c])
    __shared__ alignas(16) __bf16 ldsB[2][KPAD][CCHUNK];   // 2 x 10 KB ([k][c])
    __shared__ float ldsSq[MBLK];                          // per-pixel sum sq

    const int t     = threadIdx.x;
    const int b     = blockIdx.x >> 7;              // 8 batches
    const int m0    = (blockIdx.x & 127) * MBLK;    // pixel tile base

    // --- loader role: 128 pixels x 2 c-halves ---
    const int mload = t & 127;
    const int cp    = t >> 7;                       // 0 or 1 (c halves of 16)

    // --- compute role: 8 waves, wave w owns rows [16w,16w+16) x 10 n-tiles ---
    const int lane  = t & 31;
    const int wave  = t >> 5;
    const int mbase = wave * 16;
    const int mrow  = mbase + (lane & 15);
    const int aoff  = (lane < 16) ? 0 : 8;          // A frag: K base per half-wave
    const int boff  = (lane < 16) ? 0 : 8;          // B frag: K base per half-wave

    v8f acc[10] = {};                               // 10 tiles of 16x16 f32
    float sq = 0.f;                                 // fused sum-of-squares

    const float* fbase = feat + (size_t)b * CDIM * HWDIM + m0 + mload;

    alignas(16) __bf16 abuf[16];
    u32x4 bbuf[3];

    // ---- register-load helpers (inlined manually via lambdas) ----
    auto loadRegs = [&](int chunk) {
        const int c0 = chunk * CCHUNK;
        #pragma unroll
        for (int i = 0; i < 16; ++i) {
            float v = fbase[(size_t)(c0 + cp * 16 + i) * HWDIM];
            sq += v * v;
            abuf[i] = (__bf16)v;                    // native cvt to bf16
        }
        #pragma unroll
        for (int u = 0; u < 3; ++u) {
            const int idx = t + u * 256;
            if (idx < (KPAD * CCHUNK) / 8) {        // 640 b128 units
                const int kr   = idx >> 2;
                const int part = idx & 3;
                bbuf[u] = *(const u32x4*)(wsP + kr * CDIM + c0 + part * 8);
            }
        }
    };
    auto storeLds = [&](int buf) {
        *(u32x4*)&ldsA[buf][mload][cp * 16]     = *(const u32x4*)&abuf[0];
        *(u32x4*)&ldsA[buf][mload][cp * 16 + 8] = *(const u32x4*)&abuf[8];
        #pragma unroll
        for (int u = 0; u < 3; ++u) {
            const int idx = t + u * 256;
            if (idx < (KPAD * CCHUNK) / 8) {
                const int kr   = idx >> 2;
                const int part = idx & 3;
                *(u32x4*)&ldsB[buf][kr][part * 8] = bbuf[u];
            }
        }
    };

    // ---- software pipeline: prefetch regs(i+1) | wmma on buf(i) ----
    loadRegs(0);
    storeLds(0);
    __syncthreads();

    for (int chunk = 0; chunk < NCHUNKS; ++chunk) {
        const int cur = chunk & 1;

        if (chunk + 1 < NCHUNKS) loadRegs(chunk + 1);

        // A 16x32 bf16: lane<16 -> m=lane,    K = {0..7, 16..23}
        //               lane>=16 -> m=lane-16, K = {8..15, 24..31}
        BF16Frag a;
        a.h[0] = *(const v8bf*)&ldsA[cur][mrow][aoff];
        a.h[1] = *(const v8bf*)&ldsA[cur][mrow][aoff + 16];

        #pragma unroll
        for (int j = 0; j < 10; ++j) {
            // B 32x16 bf16: lane<16 -> n=lane,    K = {0..7, 16..23}
            //               lane>=16 -> n=lane-16, K = {8..15, 24..31}
            const int kr = j * 16 + (lane & 15);
            BF16Frag bm;
            bm.h[0] = *(const v8bf*)&ldsB[cur][kr][boff];
            bm.h[1] = *(const v8bf*)&ldsB[cur][kr][boff + 16];
            acc[j] = __builtin_amdgcn_wmma_f32_16x16x32_bf16(
                false, a.full, false, bm.full, (short)0, acc[j], false, false);
        }

        if (chunk + 1 < NCHUNKS) {
            storeLds(1 - cur);        // opposite parity: consumed & barriered
            __syncthreads();
        }
    }

    // ---- reduce fused sum-of-squares: two c-halves per pixel ----
    __syncthreads();
    if (cp == 1) ldsSq[mload] = sq;
    __syncthreads();
    if (cp == 0) ldsSq[mload] = sq + ldsSq[mload];
    __syncthreads();

    // ---- per-lane feature norms for the 8 C/D rows it owns ----
    // C/D 16x16 f32 layout: VGPR r -> row r (lane<16) / row 8+r (lane>=16)
    const int msel = mbase + ((lane < 16) ? 0 : 8);
    float fn[8];
    #pragma unroll
    for (int r = 0; r < 8; ++r) fn[r] = sqrtf(ldsSq[msel + r]);

    // ---- normalize + store (8 consecutive pixels per lane -> 2x b128) ----
    float* obase = out + (size_t)b * KOUT * HWDIM + m0 + msel;
    #pragma unroll
    for (int j = 0; j < 10; ++j) {
        const int kc = j * 16 + (lane & 15);
        if (kc < KOUT) {
            const float pn = wsN[kc];
            float v[8];
            #pragma unroll
            for (int r = 0; r < 8; ++r)
                v[r] = acc[j][r] / fmaxf(fn[r] * pn, EPS);
            f32x4 s0 = { v[0], v[1], v[2], v[3] };
            f32x4 s1 = { v[4], v[5], v[6], v[7] };
            float* p = obase + (size_t)kc * HWDIM;
            *(f32x4*)p       = s0;
            *(f32x4*)(p + 4) = s1;
        }
    }
}

// ---------------------------------------------------------------------------
extern "C" void kernel_launch(void* const* d_in, const int* in_sizes, int n_in,
                              void* d_out, int out_size, void* d_ws, size_t ws_size,
                              hipStream_t stream) {
    (void)in_sizes; (void)n_in; (void)out_size; (void)ws_size;
    const float* feat  = (const float*)d_in[0];
    const float* proto = (const float*)d_in[1];

    __bf16* wsP = (__bf16*)d_ws;                                   // 160*512 bf16
    float*  wsN = (float*)((char*)d_ws + KPAD * CDIM * 2);         // 160 f32

    float* out = (float*)d_out;

    proto_prep<<<KPAD, 128, 0, stream>>>(proto, wsP, wsN);
    cos_sim_gemm<<<BATCH * (HWDIM / MBLK), 256, 0, stream>>>(feat, wsP, wsN, out);
}
